// Transformer_46952582480472
// MI455X (gfx1250) — compile-verified, matrix operand order checked by reading, not run on hardware
//
#include <hip/hip_runtime.h>
#include <hip/hip_bf16.h>

typedef __bf16 bf16_t;
typedef __attribute__((ext_vector_type(16))) __bf16 v16bf;
typedef __attribute__((ext_vector_type(8)))  float  v8f;

#define kB    2
#define kS    1024
#define kD    512
#define kH    8
#define kL    6
#define kDK   64
#define kDFF  2048
#define kV    32000
#define kMS1  1025
#define kEps  1e-6f

union Frag { v16bf v; uint4 q[2]; };

// ---------------------------------------------------------------------------
// x[b,s,:] = emb[inp[b,s]] * sqrt(D) + PE(s,:)
// ---------------------------------------------------------------------------
__global__ __launch_bounds__(256)
void embed_pe_kernel(const int* __restrict__ inp, const float* __restrict__ emb,
                     float* __restrict__ x) {
  int row = blockIdx.x;                 // b*S + s
  int s   = row % kS;
  int tok = inp[row];
  const float sc = 22.627416997969522f; // sqrt(512)
  const float c  = -__logf(10000.0f) / (float)kD;
  for (int d = threadIdx.x; d < kD; d += blockDim.x) {
    float ang = (float)s * __expf((float)(d & ~1) * c);
    float pe  = (d & 1) ? __cosf(ang) : __sinf(ang);
    x[(size_t)row * kD + d] = emb[(size_t)tok * kD + d] * sc + pe;
  }
}

// ---------------------------------------------------------------------------
// LayerNorm (ddof=1) -> bf16 output (feeds WMMA GEMMs)
// ---------------------------------------------------------------------------
__global__ __launch_bounds__(256)
void layernorm_bf16_kernel(const float* __restrict__ x, const float* __restrict__ a,
                           const float* __restrict__ b, bf16_t* __restrict__ out) {
  __shared__ float s1[8], s2[8];
  int row = blockIdx.x;
  const float* xr = x + (size_t)row * kD;
  float sum = 0.f, sq = 0.f;
  for (int d = threadIdx.x; d < kD; d += blockDim.x) {
    float v = xr[d]; sum += v; sq += v * v;
  }
  for (int off = 16; off; off >>= 1) {
    sum += __shfl_down(sum, off, 32);
    sq  += __shfl_down(sq,  off, 32);
  }
  int lane = threadIdx.x & 31, wave = threadIdx.x >> 5;
  if (lane == 0) { s1[wave] = sum; s2[wave] = sq; }
  __syncthreads();
  if (threadIdx.x == 0) {
    float ts = 0.f, tq = 0.f;
    for (int w = 0; w < 8; ++w) { ts += s1[w]; tq += s2[w]; }
    s1[0] = ts; s2[0] = tq;
  }
  __syncthreads();
  float mean = s1[0] / (float)kD;
  float var  = (s2[0] - (float)kD * mean * mean) / (float)(kD - 1);
  float inv  = 1.0f / (sqrtf(var) + kEps);
  for (int d = threadIdx.x; d < kD; d += blockDim.x) {
    float v = (xr[d] - mean) * inv * a[d] + b[d];
    out[(size_t)row * kD + d] = (bf16_t)v;
  }
}

// ---------------------------------------------------------------------------
__global__ __launch_bounds__(256)
void f32_to_bf16_kernel(const float* __restrict__ in, bf16_t* __restrict__ out, long n) {
  long i = (long)blockIdx.x * blockDim.x + threadIdx.x;
  long st = (long)gridDim.x * blockDim.x;
  for (; i < n; i += st) out[i] = (bf16_t)in[i];
}

// ---------------------------------------------------------------------------
// vt[((b*H+h)*DK+dk)*S + s] = v[(b*S+s)*D + h*DK + dk]
// (makes P@V a pure "tn" WMMA GEMM with b128 fragment loads)
// ---------------------------------------------------------------------------
__global__ __launch_bounds__(256)
void transpose_v_kernel(const bf16_t* __restrict__ v, bf16_t* __restrict__ vt) {
  long i = (long)blockIdx.x * blockDim.x + threadIdx.x;
  long n = (long)kB * kS * kD;
  if (i >= n) return;
  int d = (int)(i % kD);
  int s = (int)((i / kD) % kS);
  int b = (int)(i / ((long)kD * kS));
  int h = d / kDK, dk = d % kDK;
  vt[(((size_t)b * kH + h) * kDK + dk) * kS + s] = v[i];
}

// ---------------------------------------------------------------------------
// idx[b,q,k] = (k + (inclusive_cumsum(mask[b,q,:])[k]-1)*nz*S) * nz
// ---------------------------------------------------------------------------
__global__ __launch_bounds__(1024)
void build_idx_kernel(const int* __restrict__ mask, int* __restrict__ idx) {
  __shared__ int sm[kS];
  int row = blockIdx.x;                 // b*S + q
  int k   = threadIdx.x;
  int m   = mask[(size_t)row * kS + k];
  sm[k] = m;
  __syncthreads();
  for (int off = 1; off < kS; off <<= 1) {
    int v = (k >= off) ? sm[k - off] : 0;
    __syncthreads();
    sm[k] += v;
    __syncthreads();
  }
  int nz = (m != 0) ? 1 : 0;
  int cs = (sm[k] - 1) * nz * kS;
  idx[(size_t)row * kS + k] = (k + cs) * nz;
}

// ---------------------------------------------------------------------------
// Fused: s = mask ? scores + rel.flat[idx] : -1e9 ; softmax ; write bf16 p
// ---------------------------------------------------------------------------
__global__ __launch_bounds__(256)
void rel_softmax_kernel(const float* __restrict__ scores, const float* __restrict__ relm,
                        const int* __restrict__ idx, const int* __restrict__ mask,
                        bf16_t* __restrict__ p) {
  __shared__ float red[8];
  int gb = blockIdx.x;                  // ((b*H)+h)*S + q
  int q  = gb % kS;
  int h  = (gb / kS) % kH;
  int b  = gb / (kS * kH);
  const float* srow = scores + (size_t)gb * kS;
  const float* rbase = relm + (size_t)(b * kH + h) * kS * kS;
  const int* irow = idx + ((size_t)b * kS + q) * kS;
  const int* mrow = mask + ((size_t)b * kS + q) * kS;
  bf16_t* prow = p + (size_t)gb * kS;

  float loc[4];
  float mx = -1e30f;
  #pragma unroll
  for (int t = 0; t < 4; ++t) {
    int k = threadIdx.x + t * 256;
    float v = (mrow[k] != 0) ? (srow[k] + rbase[irow[k]]) : -1.0e9f;
    loc[t] = v;
    mx = fmaxf(mx, v);
  }
  for (int off = 16; off; off >>= 1) mx = fmaxf(mx, __shfl_down(mx, off, 32));
  int lane = threadIdx.x & 31, wave = threadIdx.x >> 5;
  if (lane == 0) red[wave] = mx;
  __syncthreads();
  if (threadIdx.x == 0) {
    float t0 = red[0];
    for (int w = 1; w < 8; ++w) t0 = fmaxf(t0, red[w]);
    red[0] = t0;
  }
  __syncthreads();
  mx = red[0];
  __syncthreads();

  float sum = 0.f;
  #pragma unroll
  for (int t = 0; t < 4; ++t) {
    float e = __expf(loc[t] - mx);
    loc[t] = e; sum += e;
  }
  for (int off = 16; off; off >>= 1) sum += __shfl_down(sum, off, 32);
  if (lane == 0) red[wave] = sum;
  __syncthreads();
  if (threadIdx.x == 0) {
    float t0 = 0.f;
    for (int w = 0; w < 8; ++w) t0 += red[w];
    red[0] = t0;
  }
  __syncthreads();
  float inv = 1.0f / red[0];
  #pragma unroll
  for (int t = 0; t < 4; ++t) {
    int k = threadIdx.x + t * 256;
    prow[k] = (bf16_t)(loc[t] * inv);
  }
}

// ---------------------------------------------------------------------------
// log_softmax in place over last dim (V)
// ---------------------------------------------------------------------------
__global__ __launch_bounds__(256)
void log_softmax_kernel(float* __restrict__ out) {
  __shared__ float red[8];
  int row = blockIdx.x;
  float* r = out + (size_t)row * kV;
  int lane = threadIdx.x & 31, wave = threadIdx.x >> 5;

  float mx = -1e30f;
  for (int k = threadIdx.x; k < kV; k += blockDim.x) mx = fmaxf(mx, r[k]);
  for (int off = 16; off; off >>= 1) mx = fmaxf(mx, __shfl_down(mx, off, 32));
  if (lane == 0) red[wave] = mx;
  __syncthreads();
  if (threadIdx.x == 0) {
    float t0 = red[0];
    for (int w = 1; w < 8; ++w) t0 = fmaxf(t0, red[w]);
    red[0] = t0;
  }
  __syncthreads();
  mx = red[0];
  __syncthreads();

  float sum = 0.f;
  for (int k = threadIdx.x; k < kV; k += blockDim.x) sum += __expf(r[k] - mx);
  for (int off = 16; off; off >>= 1) sum += __shfl_down(sum, off, 32);
  if (lane == 0) red[wave] = sum;
  __syncthreads();
  if (threadIdx.x == 0) {
    float t0 = 0.f;
    for (int w = 0; w < 8; ++w) t0 += red[w];
    red[0] = t0;
  }
  __syncthreads();
  float lse = mx + __logf(red[0]);
  for (int k = threadIdx.x; k < kV; k += blockDim.x) r[k] -= lse;
}

// ---------------------------------------------------------------------------
// WMMA GEMM "tn": C[m,n] = alpha * sum_k A[m,k]*B[n,k] (+bias)(+add)(ReLU?)
// A and B K-major (row strides lda/ldb). Each wave computes a 32x32 block
// (2 M-tiles x 2 N-tiles): 8 b128 loads per 4 WMMA (2:1), fragments fully
// shared across the 2x2 accumulator grid.
// 4 waves per block along M => block covers 128(M) x 32(N).
// Batched via z = zo*innerCount + zi with separate outer/inner strides.
// ---------------------------------------------------------------------------
template <bool RELU, bool OUT_BF16>
__global__ __launch_bounds__(128)
void gemm_tn_kernel(const bf16_t* __restrict__ A, const bf16_t* __restrict__ Bm,
                    const float* __restrict__ bias, const float* __restrict__ add,
                    float* __restrict__ Cf, bf16_t* __restrict__ Cb,
                    int K, int lda, int ldb, int ldc,
                    long aOuter, long aInner, long bOuter, long bInner,
                    long cOuter, long cInner, int innerCount, float alpha) {
  int z = blockIdx.z;
  int zo = z / innerCount, zi = z % innerCount;
  A  += (size_t)zo * aOuter + (size_t)zi * aInner;
  Bm += (size_t)zo * bOuter + (size_t)zi * bInner;
  long coff = (long)zo * cOuter + (long)zi * cInner;

  int lane = threadIdx.x & 31;
  int wave = threadIdx.x >> 5;
  int r    = lane & 15;
  int half = lane >> 4;
  int row0 = (blockIdx.y * 4 + wave) * 32;
  int col0 = blockIdx.x * 32;

  const bf16_t* ap0 = A  + (size_t)(row0 + r) * lda + half * 8;
  const bf16_t* ap1 = ap0 + (size_t)16 * lda;
  const bf16_t* bp0 = Bm + (size_t)(col0 + r) * ldb + half * 8;
  const bf16_t* bp1 = bp0 + (size_t)16 * ldb;

  v8f acc00 = {}, acc01 = {}, acc10 = {}, acc11 = {};
  for (int k0 = 0; k0 < K; k0 += 32) {
    Frag fa0, fa1, fb0, fb1;
    fa0.q[0] = *(const uint4*)(ap0);
    fa0.q[1] = *(const uint4*)(ap0 + 16);
    fa1.q[0] = *(const uint4*)(ap1);
    fa1.q[1] = *(const uint4*)(ap1 + 16);
    fb0.q[0] = *(const uint4*)(bp0);
    fb0.q[1] = *(const uint4*)(bp0 + 16);
    fb1.q[0] = *(const uint4*)(bp1);
    fb1.q[1] = *(const uint4*)(bp1 + 16);
    __builtin_prefetch(ap0 + 64, 0, 1);
    __builtin_prefetch(ap1 + 64, 0, 1);
    __builtin_prefetch(bp0 + 64, 0, 1);
    __builtin_prefetch(bp1 + 64, 0, 1);
    ap0 += 32; ap1 += 32; bp0 += 32; bp1 += 32;
    acc00 = __builtin_amdgcn_wmma_f32_16x16x32_bf16(false, fa0.v, false, fb0.v,
                                                    (short)0, acc00, false, false);
    acc01 = __builtin_amdgcn_wmma_f32_16x16x32_bf16(false, fa0.v, false, fb1.v,
                                                    (short)0, acc01, false, false);
    acc10 = __builtin_amdgcn_wmma_f32_16x16x32_bf16(false, fa1.v, false, fb0.v,
                                                    (short)0, acc10, false, false);
    acc11 = __builtin_amdgcn_wmma_f32_16x16x32_bf16(false, fa1.v, false, fb1.v,
                                                    (short)0, acc11, false, false);
  }

  int orow0 = row0 + half * 8;
  int orow1 = row0 + 16 + half * 8;
  int ocol0 = col0 + r;
  int ocol1 = col0 + 16 + r;
  float bv0 = bias ? bias[ocol0] : 0.0f;
  float bv1 = bias ? bias[ocol1] : 0.0f;
  #pragma unroll
  for (int i = 0; i < 8; ++i) {
    long o00 = coff + (long)(orow0 + i) * ldc + ocol0;
    long o01 = coff + (long)(orow0 + i) * ldc + ocol1;
    long o10 = coff + (long)(orow1 + i) * ldc + ocol0;
    long o11 = coff + (long)(orow1 + i) * ldc + ocol1;
    float v00 = acc00[i] * alpha + bv0;
    float v01 = acc01[i] * alpha + bv1;
    float v10 = acc10[i] * alpha + bv0;
    float v11 = acc11[i] * alpha + bv1;
    if (add) { v00 += add[o00]; v01 += add[o01]; v10 += add[o10]; v11 += add[o11]; }
    if (RELU) {
      v00 = fmaxf(v00, 0.0f); v01 = fmaxf(v01, 0.0f);
      v10 = fmaxf(v10, 0.0f); v11 = fmaxf(v11, 0.0f);
    }
    if (OUT_BF16) {
      Cb[o00] = (bf16_t)v00; Cb[o01] = (bf16_t)v01;
      Cb[o10] = (bf16_t)v10; Cb[o11] = (bf16_t)v11;
    } else {
      Cf[o00] = v00; Cf[o01] = v01; Cf[o10] = v10; Cf[o11] = v11;
    }
  }
}

// ---------------------------------------------------------------------------
extern "C" void kernel_launch(void* const* d_in, const int* in_sizes, int n_in,
                              void* d_out, int out_size, void* d_ws, size_t ws_size,
                              hipStream_t stream) {
  (void)in_sizes; (void)n_in; (void)out_size; (void)ws_size;
  const int*   inp  = (const int*)d_in[0];
  const int*   mask = (const int*)d_in[1];
  const float* emb  = (const float*)d_in[2];
  const float* Wq   = (const float*)d_in[3];
  const float* bq   = (const float*)d_in[4];
  const float* Wk   = (const float*)d_in[5];
  const float* bk   = (const float*)d_in[6];
  const float* Wv   = (const float*)d_in[7];
  const float* bv   = (const float*)d_in[8];
  const float* Wo   = (const float*)d_in[9];
  const float* bo   = (const float*)d_in[10];
  const float* rel  = (const float*)d_in[11];
  const float* ln1a = (const float*)d_in[12];
  const float* ln1b = (const float*)d_in[13];
  const float* ln2a = (const float*)d_in[14];
  const float* ln2b = (const float*)d_in[15];
  const float* W1   = (const float*)d_in[16];
  const float* b1   = (const float*)d_in[17];
  const float* W2   = (const float*)d_in[18];
  const float* b2   = (const float*)d_in[19];
  const float* lnfa = (const float*)d_in[20];
  const float* lnfb = (const float*)d_in[21];
  const float* Wg   = (const float*)d_in[22];
  const float* bg   = (const float*)d_in[23];
  float* out = (float*)d_out;

  // ---- workspace carve-out ----
  char* p = (char*)d_ws;
  auto alloc = [&](size_t bytes) -> void* {
    void* r = (void*)p;
    p += (bytes + 255) & ~(size_t)255;
    return r;
  };
  const size_t NT = (size_t)kB * kS;              // 2048 tokens
  float*  x0     = (float*)alloc(NT * kD * 4);
  float*  x1     = (float*)alloc(NT * kD * 4);
  bf16_t* hbf    = (bf16_t*)alloc(NT * kD * 2);
  bf16_t* qbf    = (bf16_t*)alloc(NT * kD * 2);
  bf16_t* kbf    = (bf16_t*)alloc(NT * kD * 2);
  bf16_t* vbf    = (bf16_t*)alloc(NT * kD * 2);
  bf16_t* vtbf   = (bf16_t*)alloc(NT * kD * 2);
  bf16_t* abf    = (bf16_t*)alloc(NT * kD * 2);
  bf16_t* f1bf   = (bf16_t*)alloc(NT * kDFF * 2);
  float*  scores = (float*)alloc((size_t)kB * kH * kS * kS * 4);
  float*  relm   = (float*)alloc((size_t)kB * kH * kS * kS * 4);
  bf16_t* pbf    = (bf16_t*)alloc((size_t)kB * kH * kS * kS * 2);
  int*    idx    = (int*)alloc((size_t)kB * kS * kS * 4);
  bf16_t* Wqb  = (bf16_t*)alloc((size_t)kL * kD * kD * 2);
  bf16_t* Wkb  = (bf16_t*)alloc((size_t)kL * kD * kD * 2);
  bf16_t* Wvb  = (bf16_t*)alloc((size_t)kL * kD * kD * 2);
  bf16_t* Wob  = (bf16_t*)alloc((size_t)kL * kD * kD * 2);
  bf16_t* W1b  = (bf16_t*)alloc((size_t)kL * kDFF * kD * 2);
  bf16_t* W2b  = (bf16_t*)alloc((size_t)kL * kD * kDFF * 2);
  bf16_t* relb = (bf16_t*)alloc((size_t)kL * kH * kMS1 * kDK * 2);
  bf16_t* Wgb  = (bf16_t*)alloc((size_t)kV * kD * 2);

  auto cvt = [&](const float* src, bf16_t* dst, long n) {
    f32_to_bf16_kernel<<<1024, 256, 0, stream>>>(src, dst, n);
  };
  cvt(Wq,  Wqb,  (long)kL * kD * kD);
  cvt(Wk,  Wkb,  (long)kL * kD * kD);
  cvt(Wv,  Wvb,  (long)kL * kD * kD);
  cvt(Wo,  Wob,  (long)kL * kD * kD);
  cvt(W1,  W1b,  (long)kL * kDFF * kD);
  cvt(W2,  W2b,  (long)kL * kD * kDFF);
  cvt(rel, relb, (long)kL * kH * kMS1 * kDK);
  cvt(Wg,  Wgb,  (long)kV * kD);

  embed_pe_kernel<<<(int)NT, 256, 0, stream>>>(inp, emb, x0);
  build_idx_kernel<<<(int)NT, 1024, 0, stream>>>(mask, idx);

  const float scale = 0.125f;           // 1/sqrt(DK)
  float* xc = x0;
  float* xn = x1;

  for (int i = 0; i < kL; ++i) {
    // ---- LN1 -> h (bf16) ----
    layernorm_bf16_kernel<<<(int)NT, 256, 0, stream>>>(xc, ln1a + i * kD, ln1b + i * kD, hbf);

    // ---- Q,K,V projections (bf16 out, layout (b,s,h,dk)) ----
    dim3 gq(kD / 32, (int)(NT / 128), 1);
    gemm_tn_kernel<false, true><<<gq, 128, 0, stream>>>(
        hbf, Wqb + (size_t)i * kD * kD, bq + i * kD, nullptr, nullptr, qbf,
        kD, kD, kD, kD, 0, 0, 0, 0, 0, 0, 1, 1.0f);
    gemm_tn_kernel<false, true><<<gq, 128, 0, stream>>>(
        hbf, Wkb + (size_t)i * kD * kD, bk + i * kD, nullptr, nullptr, kbf,
        kD, kD, kD, kD, 0, 0, 0, 0, 0, 0, 1, 1.0f);
    gemm_tn_kernel<false, true><<<gq, 128, 0, stream>>>(
        hbf, Wvb + (size_t)i * kD * kD, bv + i * kD, nullptr, nullptr, vbf,
        kD, kD, kD, kD, 0, 0, 0, 0, 0, 0, 1, 1.0f);

    // ---- V -> (b,h,dk,s) for the PV GEMM ----
    transpose_v_kernel<<<(int)((NT * kD + 255) / 256), 256, 0, stream>>>(vbf, vtbf);

    // ---- scores = scale * Q K^T  (batched over b,h) ----
    dim3 gs(kS / 32, kS / 128, kB * kH);
    gemm_tn_kernel<false, false><<<gs, 128, 0, stream>>>(
        qbf, kbf, nullptr, nullptr, scores, nullptr,
        kDK, kD, kD, kS,
        (long)kS * kD, (long)kDK, (long)kS * kD, (long)kDK,
        (long)kH * kS * kS, (long)kS * kS, kH, scale);

    // ---- relm[m,l] = scale * PE[h,m,:] . Q[b,h,l,:] ----
    gemm_tn_kernel<false, false><<<gs, 128, 0, stream>>>(
        relb + (size_t)i * kH * kMS1 * kDK, qbf, nullptr, nullptr, relm, nullptr,
        kDK, kDK, kD, kS,
        0L, (long)kMS1 * kDK, (long)kS * kD, (long)kDK,
        (long)kH * kS * kS, (long)kS * kS, kH, scale);

    // ---- gather rel, mask, softmax -> p (bf16) ----
    rel_softmax_kernel<<<kB * kH * kS, 256, 0, stream>>>(scores, relm, idx, mask, pbf);

    // ---- a = P @ V : tn GEMM vs transposed V (bf16 out, (b,q,h*dk)) ----
    dim3 gpv(kDK / 32, kS / 128, kB * kH);
    gemm_tn_kernel<false, true><<<gpv, 128, 0, stream>>>(
        pbf, vtbf, nullptr, nullptr, nullptr, abf,
        kS, kS, kS, kD,
        (long)kH * kS * kS, (long)kS * kS, (long)kH * kDK * kS, (long)kDK * kS,
        (long)kS * kD, (long)kDK, kH, 1.0f);

    // ---- x = x + a Wo^T + bo ----
    gemm_tn_kernel<false, false><<<gq, 128, 0, stream>>>(
        abf, Wob + (size_t)i * kD * kD, bo + i * kD, xc, xn, nullptr,
        kD, kD, kD, kD, 0, 0, 0, 0, 0, 0, 1, 1.0f);
    { float* t = xc; xc = xn; xn = t; }

    // ---- LN2 -> h2 (bf16) ----
    layernorm_bf16_kernel<<<(int)NT, 256, 0, stream>>>(xc, ln2a + i * kD, ln2b + i * kD, hbf);

    // ---- f1 = relu(h2 W1^T + b1) (bf16) ----
    dim3 g1(kDFF / 32, (int)(NT / 128), 1);
    gemm_tn_kernel<true, true><<<g1, 128, 0, stream>>>(
        hbf, W1b + (size_t)i * kDFF * kD, b1 + i * kDFF, nullptr, nullptr, f1bf,
        kD, kD, kD, kDFF, 0, 0, 0, 0, 0, 0, 1, 1.0f);

    // ---- x = x + f1 W2^T + b2 ----
    gemm_tn_kernel<false, false><<<gq, 128, 0, stream>>>(
        f1bf, W2b + (size_t)i * kD * kDFF, b2 + i * kD, xc, xn, nullptr,
        kDFF, kDFF, kDFF, kD, 0, 0, 0, 0, 0, 0, 1, 1.0f);
    { float* t = xc; xc = xn; xn = t; }
  }

  // ---- final LN -> vocab projection -> log_softmax ----
  layernorm_bf16_kernel<<<(int)NT, 256, 0, stream>>>(xc, lnfa, lnfb, hbf);
  dim3 gg(kV / 32, (int)(NT / 128), 1);
  gemm_tn_kernel<false, false><<<gg, 128, 0, stream>>>(
      hbf, Wgb, bg, nullptr, out, nullptr,
      kD, kD, kD, kV, 0, 0, 0, 0, 0, 0, 1, 1.0f);
  log_softmax_kernel<<<(int)NT, 256, 0, stream>>>(out);
}
